// MinkEncoder_56470230008551
// MI455X (gfx1250) — matrix operand order repaired and checked by reading, not compiled, not run on hardware
//
#include <hip/hip_runtime.h>

typedef __attribute__((ext_vector_type(16))) _Float16 v16h;
typedef __attribute__((ext_vector_type(8)))  float    v8f;

// ---------------- problem dimensions ----------------
constexpr int BATCH = 2;
constexpr int D0=32, H0=192, W0=192;   // level 0
constexpr int D1=17, H1=97,  W1=97;    // after stride-2 k2 p1
constexpr int D2=9,  H2=49,  W2=49;
constexpr int D3=5,  H3=25,  W3=25;
constexpr int COUT_FINAL = 256;

constexpr size_t N0 = (size_t)BATCH*D0*H0*W0;   // 2,359,296
constexpr size_t N1 = (size_t)BATCH*D1*H1*W1;   // 319,906
constexpr size_t N2 = (size_t)BATCH*D2*H2*W2;   // 43,218
constexpr size_t N3 = (size_t)BATCH*D3*H3*W3;   // 6,250
constexpr float  EPS = 1e-5f;

// ---------------- static device pool (all regions 16-float aligned for b128) ----
constexpr size_t AL(size_t x) { return (x + 15) & ~(size_t)15; }
constexpr size_t OFF_X0   = 0;                        // dense input, 4ch
constexpr size_t OFF_M0   = OFF_X0 + N0*4;            // mask L0 (contiguous for one-shot zero)
constexpr size_t OFF_T0A  = AL(OFF_M0 + N0);          // 16ch L0
constexpr size_t OFF_T0B  = AL(OFF_T0A + N0*16);      // 16ch L0
constexpr size_t OFF_M1   = AL(OFF_T0B + N0*16);
constexpr size_t OFF_T1A  = AL(OFF_M1  + N1);         // 32ch-capacity L1 buffers
constexpr size_t OFF_T1B  = AL(OFF_T1A + N1*32);
constexpr size_t OFF_T1C  = AL(OFF_T1B + N1*32);
constexpr size_t OFF_T1D  = AL(OFF_T1C + N1*32);
constexpr size_t OFF_M2   = AL(OFF_T1D + N1*32);
constexpr size_t OFF_T2A  = AL(OFF_M2  + N2);         // 64ch-capacity L2 buffers
constexpr size_t OFF_T2B  = AL(OFF_T2A + N2*64);
constexpr size_t OFF_T2C  = AL(OFF_T2B + N2*64);
constexpr size_t OFF_T2D  = AL(OFF_T2C + N2*64);
constexpr size_t OFF_M3   = AL(OFF_T2D + N2*64);
constexpr size_t OFF_T3A  = AL(OFF_M3  + N3);         // 128ch-capacity L3 buffers
constexpr size_t OFF_T3B  = AL(OFF_T3A + N3*128);
constexpr size_t OFF_T3C  = AL(OFF_T3B + N3*128);
constexpr size_t OFF_T3D  = AL(OFF_T3C + N3*128);     // reused as repacked final acts
constexpr size_t OFF_STATS= AL(OFF_T3D + N3*128);     // [0..127] sum, [128..255] sumsq
constexpr size_t OFF_NS   = OFF_STATS + 256;          // n per level (4 scalars)
constexpr size_t G_TOTAL  = OFF_NS + 4;

__device__ __attribute__((aligned(256))) float G[G_TOTAL];

// Weight fragments prepacked into per-lane WMMA B order (f16).
// Max: 8 cout-tiles * 108 ksteps * 32 lanes * 16 halves = 442,368
constexpr size_t WPACK_MAX = 512*1024;
__device__ __attribute__((aligned(256))) _Float16 WPACK[WPACK_MAX];

// ---------------- elementwise / setup kernels ----------------
__global__ void k_zero4(size_t off, size_t n4) {   // n4 = count of float4
  size_t i = (size_t)blockIdx.x*blockDim.x + threadIdx.x;
  size_t st = (size_t)gridDim.x*blockDim.x;
  float4 z = make_float4(0.f,0.f,0.f,0.f);
  for (; i < n4; i += st) *(float4*)&G[off + i*4] = z;
}

__global__ void k_scatter(const float* __restrict__ feats,
                          const int* __restrict__ coors, int nvox) {
  int i = blockIdx.x*blockDim.x + threadIdx.x;
  if (i >= nvox) return;
  int b = coors[4*i+0], z = coors[4*i+1], y = coors[4*i+2], x = coors[4*i+3];
  size_t v = (((size_t)b*D0 + z)*H0 + y)*W0 + x;
  float4 f = *(const float4*)&feats[4*i];
  *(float4*)&G[OFF_X0 + v*4] = f;
  G[OFF_M0 + v] = 1.0f;
}

__global__ void k_masksum(size_t moff, int n, size_t nsoff) {
  __shared__ float sh[1024];
  float s = 0.f;
  for (int i = threadIdx.x; i < n; i += 1024) s += G[moff+i];
  sh[threadIdx.x] = s; __syncthreads();
  for (int o = 512; o > 0; o >>= 1) {
    if (threadIdx.x < (unsigned)o) sh[threadIdx.x] += sh[threadIdx.x+o];
    __syncthreads();
  }
  if (threadIdx.x == 0) G[nsoff] = fmaxf(sh[0], 1.0f);
}

__global__ void k_downmask(size_t inoff, int Din,int Hin,int Win,
                           size_t outoff, int Dout,int Hout,int Wout) {
  int total = BATCH*Dout*Hout*Wout;
  int i = blockIdx.x*blockDim.x + threadIdx.x;
  if (i >= total) return;
  int t = i;
  int x = t % Wout; t /= Wout;
  int y = t % Hout; t /= Hout;
  int z = t % Dout; t /= Dout;
  int b = t;
  float mx = 0.f;
  #pragma unroll
  for (int dz = 0; dz < 2; ++dz)
  for (int dy = 0; dy < 2; ++dy)
  for (int dx = 0; dx < 2; ++dx) {
    int zi = 2*z - 1 + dz, yi = 2*y - 1 + dy, xi = 2*x - 1 + dx;
    if (zi>=0 && zi<Din && yi>=0 && yi<Hin && xi>=0 && xi<Win)
      mx = fmaxf(mx, G[inoff + (((size_t)b*Din + zi)*Hin + yi)*Win + xi]);
  }
  G[outoff + i] = mx;
}

// per-channel masked sum / sumsq (deterministic: fixed reduction tree)
__global__ void k_stats(size_t toff, size_t moff, int nvox, int C) {
  int c = blockIdx.x;
  __shared__ float s1[1024], s2[1024];
  float s = 0.f, ss = 0.f;
  for (int v = threadIdx.x; v < nvox; v += 1024) {
    float x = G[toff + (size_t)v*C + c] * G[moff + v];
    s += x; ss += x*x;
  }
  s1[threadIdx.x] = s; s2[threadIdx.x] = ss; __syncthreads();
  for (int o = 512; o > 0; o >>= 1) {
    if (threadIdx.x < (unsigned)o) { s1[threadIdx.x]+=s1[threadIdx.x+o]; s2[threadIdx.x]+=s2[threadIdx.x+o]; }
    __syncthreads();
  }
  if (threadIdx.x == 0) { G[OFF_STATS + c] = s1[0]; G[OFF_STATS + 128 + c] = s2[0]; }
}

// vectorized: 4 consecutive channels of one voxel per thread (C % 4 == 0)
__global__ void k_bn(size_t inoff, size_t outoff, size_t moff, size_t nsoff,
                     const float* __restrict__ gma, const float* __restrict__ bta,
                     int nvox, int C, int relu) {
  const int C4 = C >> 2;
  size_t total4 = (size_t)nvox*C4;
  size_t i = (size_t)blockIdx.x*blockDim.x + threadIdx.x;
  if (i >= total4) return;
  int c = (int)(i % C4) * 4;
  size_t v = i / C4;
  float n  = G[nsoff];
  float mk = G[moff + v];
  float4 x = *(const float4*)&G[inoff + i*4];
  float4 y;
  float* xp = &x.x; float* yp = &y.x;
  #pragma unroll
  for (int j = 0; j < 4; ++j) {
    float mean = G[OFF_STATS + c + j] / n;
    float var  = G[OFF_STATS + 128 + c + j] / n - mean*mean;
    float rs   = rsqrtf(var + EPS);
    float xv   = xp[j] * mk;
    float o    = ((xv - mean) * rs * gma[c+j] + bta[c+j]) * mk;
    yp[j] = relu ? fmaxf(o, 0.0f) : o;
  }
  *(float4*)&G[outoff + i*4] = y;
}

__global__ void k_addrelu4(size_t aoff, size_t boff, size_t ooff, size_t n4) {
  size_t i = (size_t)blockIdx.x*blockDim.x + threadIdx.x;
  size_t st = (size_t)gridDim.x*blockDim.x;
  for (; i < n4; i += st) {
    float4 a = *(const float4*)&G[aoff + i*4];
    float4 b = *(const float4*)&G[boff + i*4];
    float4 o;
    o.x = fmaxf(a.x+b.x, 0.f); o.y = fmaxf(a.y+b.y, 0.f);
    o.z = fmaxf(a.z+b.z, 0.f); o.w = fmaxf(a.w+b.w, 0.f);
    *(float4*)&G[ooff + i*4] = o;
  }
}

// ---------------- weight prepack (conv layout: w[k][Cout]) ----------------
// WPACK[((ct*ksteps + step)*32 + lane)*16 + h] holds
// B(K = step*32 + (lane>=16?8:0) + h + (h>=8?8:0), N = ct*16 + (lane&15)).
__global__ void k_pack_w(const float* __restrict__ w, int Cout, int R, int ksteps) {
  const int ct = blockIdx.x, step = blockIdx.y, lane = threadIdx.x;
  const int n = ct*16 + (lane & 15);
  const int khi = (lane >= 16) ? 8 : 0;
  const size_t base = (((size_t)ct*ksteps + step)*32 + lane)*16;
  #pragma unroll
  for (int h = 0; h < 16; ++h) {
    int k = step*32 + khi + h + (h >= 8 ? 8 : 0);
    float v = (k < R) ? w[(size_t)k*Cout + n] : 0.f;
    WPACK[base + h] = (_Float16)v;
  }
}

// final weights layout: fw[o][k] (o-major) -> same fragment order
__global__ void k_pack_final(const float* __restrict__ fw, int R, int ksteps) {
  const int ct = blockIdx.x, step = blockIdx.y, lane = threadIdx.x;
  const int n = ct*16 + (lane & 15);
  const int khi = (lane >= 16) ? 8 : 0;
  const size_t base = (((size_t)ct*ksteps + step)*32 + lane)*16;
  #pragma unroll
  for (int h = 0; h < 16; ++h) {
    int k = step*32 + khi + h + (h >= 8 ? 8 : 0);
    float v = (k < R) ? fw[(size_t)n*R + k] : 0.f;
    WPACK[base + h] = (_Float16)v;
  }
}

// ---------------- implicit-GEMM 3D conv via WMMA ----------------
// Each wave: one 16-voxel M-tile x NT cout-tiles, one A fragment reused by NT
// back-to-back WMMAs per K-step (cuts A traffic NT-fold on deep layers).
// A layout: lane e[0..7] = K=(hi?8:0)+h, e[8..15] = K=16+(hi?8:0)+h -> two runs of
// 8 consecutive reduction indices = 8 contiguous 32B-aligned floats (CIN%8==0).
// B fragments come from WPACK via pointer induction (one VGPR-pair per nt,
// bumped by a constant stride each step -> no per-access base rematerialization).
template<int CIN, int KS, int STRIDE, int PAD, int NT>
__global__ void k_conv_wmma(size_t inoff, size_t outoff, int Cout,
                            int Din, int Hin, int Win,
                            int Dout, int Hout, int Wout) {
  constexpr int R = KS*KS*KS*CIN;
  constexpr int KSTEPS = (R + 31) / 32;
  constexpr int FRAG = 32*16;                 // halves per (step) per cout-tile
  const int lane = threadIdx.x & 31;
  const int wave = threadIdx.x >> 5;
  const int nvox = BATCH*Dout*Hout*Wout;
  const int tile = (blockIdx.x*8 + wave) * 16;
  const int ct0 = blockIdx.y * NT;
  const bool hi = lane >= 16;
  const int  m  = lane & 15;

  int vox = tile + m; if (vox >= nvox) vox = 0;       // clamp; stores are guarded
  int t = vox;
  const int xo = t % Wout; t /= Wout;
  const int yo = t % Hout; t /= Hout;
  const int zo = t % Dout; t /= Dout;
  const int b  = t;

  const float* __restrict__ inp = &G[inoff];
  const _Float16* wpn[NT];
  #pragma unroll
  for (int nt = 0; nt < NT; ++nt)
    wpn[nt] = &WPACK[(((size_t)(ct0 + nt)*KSTEPS)*32 + lane)*16];

  v8f acc[NT];
  #pragma unroll
  for (int nt = 0; nt < NT; ++nt) acc[nt] = v8f{};

  for (int step = 0; step < KSTEPS; ++step) {
    const int kbase = step*32;
    union { v16h v; _Float16 e[16]; } a;
    if constexpr ((CIN & 7) == 0) {
      #pragma unroll
      for (int r = 0; r < 2; ++r) {
        const int k0 = kbase + r*16 + (hi ? 8 : 0);
        float va[8] = {0.f,0.f,0.f,0.f,0.f,0.f,0.f,0.f};
        if (k0 < R) {
          const int tap = k0 / CIN;            // power-of-two -> shift
          const int ci  = k0 % CIN;
          const int kx = tap % KS, ky = (tap/KS) % KS, kz = tap/(KS*KS);
          const int zi = zo*STRIDE + kz - PAD;
          const int yi = yo*STRIDE + ky - PAD;
          const int xi = xo*STRIDE + kx - PAD;
          if (zi>=0 && zi<Din && yi>=0 && yi<Hin && xi>=0 && xi<Win) {
            const float4* p = (const float4*)(inp +
                ((((size_t)b*Din + zi)*Hin + yi)*Win + xi)*CIN + ci);
            float4 f0 = p[0], f1 = p[1];
            va[0]=f0.x; va[1]=f0.y; va[2]=f0.z; va[3]=f0.w;
            va[4]=f1.x; va[5]=f1.y; va[6]=f1.z; va[7]=f1.w;
          }
        }
        #pragma unroll
        for (int j = 0; j < 8; ++j) a.e[r*8 + j] = (_Float16)va[j];
      }
    } else {
      #pragma unroll
      for (int h = 0; h < 16; ++h) {
        const int v4 = h >> 1, half = h & 1;
        const int Ka = ((v4 < 4) ? 0 : 16) + (hi ? 8 : 0) + 2*(v4 & 3) + half;
        const int k = kbase + Ka;
        float av = 0.f;
        if (k < R) {
          const int ci = k % CIN, kk = k / CIN;
          const int kx = kk % KS, ky = (kk/KS) % KS, kz = kk/(KS*KS);
          const int zi = zo*STRIDE + kz - PAD;
          const int yi = yo*STRIDE + ky - PAD;
          const int xi = xo*STRIDE + kx - PAD;
          if (zi>=0 && zi<Din && yi>=0 && yi<Hin && xi>=0 && xi<Win)
            av = inp[((((size_t)b*Din + zi)*Hin + yi)*Win + xi)*CIN + ci];
        }
        a.e[h] = (_Float16)av;
      }
    }
    #pragma unroll
    for (int nt = 0; nt < NT; ++nt) {
      v16h bw = *(const v16h*)wpn[nt];
      wpn[nt] += FRAG;
      acc[nt] = __builtin_amdgcn_wmma_f32_16x16x32_f16(false, a.v, false, bw,
                                                       (short)0, acc[nt], false, false);
    }
  }

  float* __restrict__ outp = &G[outoff];
  #pragma unroll
  for (int nt = 0; nt < NT; ++nt) {
    union { v8f v; float e[8]; } o; o.v = acc[nt];
    #pragma unroll
    for (int j = 0; j < 8; ++j) {
      int M = j + (hi ? 8 : 0);
      int vo = tile + M;
      if (vo < nvox) outp[(size_t)vo*Cout + (ct0 + nt)*16 + m] = o.e[j];
    }
  }
}

// repack stage-3 output x[b,d,h,w,c] -> xr[(b,h,w)][c*D3 + d] for contiguous K
__global__ void k_repack_final(size_t xoff, size_t xroff) {
  const int total = (int)(BATCH*H3*W3) * 640;
  int i = blockIdx.x*blockDim.x + threadIdx.x;
  if (i >= total) return;
  int cf = i % 640;
  int sp = i / 640;
  int w = sp % W3; sp /= W3;
  int h = sp % H3; sp /= H3;
  int b = sp;
  int cq = cf / D3, dq = cf % D3;
  G[xroff + i] = G[xoff + ((((size_t)b*D3 + dq)*H3 + h)*W3 + w)*128 + cq];
}

// ---------------- final einsum (K = 640) + bias, NT = 4 cout tiles / wave ------
__global__ void k_final_wmma(size_t xroff, const float* __restrict__ fb,
                             float* __restrict__ out) {
  constexpr int R = 128*D3;                   // 640
  constexpr int KSTEPS = R/32;                // 20
  constexpr int NT = 4;
  constexpr int FRAG = 32*16;
  const int lane = threadIdx.x & 31;
  const int wave = threadIdx.x >> 5;
  const int NM = BATCH*H3*W3;                 // 1250
  const int tile = (blockIdx.x*8 + wave) * 16;
  const int ct0 = blockIdx.y * NT;
  const bool hi = lane >= 16;
  const int  m  = lane & 15;

  int mv = tile + m; if (mv >= NM) mv = 0;
  const float* __restrict__ xrow = &G[xroff + (size_t)mv*R];
  const _Float16* wpn[NT];
  #pragma unroll
  for (int nt = 0; nt < NT; ++nt)
    wpn[nt] = &WPACK[(((size_t)(ct0 + nt)*KSTEPS)*32 + lane)*16];

  v8f acc[NT];
  #pragma unroll
  for (int nt = 0; nt < NT; ++nt) acc[nt] = v8f{};

  for (int step = 0; step < KSTEPS; ++step) {
    union { v16h v; _Float16 e[16]; } a;
    #pragma unroll
    for (int r = 0; r < 2; ++r) {
      const int k0 = step*32 + r*16 + (hi ? 8 : 0);
      const float4* p = (const float4*)(xrow + k0);
      float4 f0 = p[0], f1 = p[1];
      a.e[r*8+0]=(_Float16)f0.x; a.e[r*8+1]=(_Float16)f0.y;
      a.e[r*8+2]=(_Float16)f0.z; a.e[r*8+3]=(_Float16)f0.w;
      a.e[r*8+4]=(_Float16)f1.x; a.e[r*8+5]=(_Float16)f1.y;
      a.e[r*8+6]=(_Float16)f1.z; a.e[r*8+7]=(_Float16)f1.w;
    }
    #pragma unroll
    for (int nt = 0; nt < NT; ++nt) {
      v16h bw = *(const v16h*)wpn[nt];
      wpn[nt] += FRAG;
      acc[nt] = __builtin_amdgcn_wmma_f32_16x16x32_f16(false, a.v, false, bw,
                                                       (short)0, acc[nt], false, false);
    }
  }
  #pragma unroll
  for (int nt = 0; nt < NT; ++nt) {
    union { v8f v; float e[8]; } o; o.v = acc[nt];
    #pragma unroll
    for (int j = 0; j < 8; ++j) {
      int M = j + (hi ? 8 : 0);
      int vo = tile + M;
      if (vo < NM) {
        int tt = vo;
        int ww = tt % W3; tt /= W3;
        int hh = tt % H3; tt /= H3;
        int bb = tt;
        out[(((size_t)bb*COUT_FINAL + (ct0+nt)*16 + m)*H3 + hh)*W3 + ww]
            = o.e[j] + fb[(ct0+nt)*16 + m];
      }
    }
  }
}

// ---------------- host orchestration ----------------
struct CBN { const float* w; const float* g; const float* b; };

static void launch_conv(hipStream_t s, size_t inoff, size_t outoff, const float* w,
                        int Cin, int Cout, int ks, int stride, int pad,
                        int Din,int Hin,int Win, int Dout,int Hout,int Wout) {
  const int R = ks*ks*ks*Cin;
  const int ksteps = (R + 31)/32;
  const int ct = Cout/16;
  k_pack_w<<<dim3(ct, ksteps), 32, 0, s>>>(w, Cout, R, ksteps);
  const int nout = BATCH*Dout*Hout*Wout;
  const int ntSel = (Cout >= 64) ? 4 : ((Cout == 32) ? 2 : 1);
  dim3 grid((nout + 127)/128, ct/ntSel);
  #define CONV_CASE(CI,KK,ST,PD,NTT) \
    if (Cin==CI && ks==KK && stride==ST && pad==PD && ntSel==NTT) { \
      k_conv_wmma<CI,KK,ST,PD,NTT><<<grid,256,0,s>>>(inoff,outoff,Cout,Din,Hin,Win,Dout,Hout,Wout); return; }
  CONV_CASE(4,  3,1,1,1)
  CONV_CASE(16, 3,1,1,1)
  CONV_CASE(16, 3,1,1,2)
  CONV_CASE(16, 2,2,1,1)
  CONV_CASE(16, 1,1,0,2)
  CONV_CASE(32, 3,1,1,2)
  CONV_CASE(32, 3,1,1,4)
  CONV_CASE(32, 2,2,1,2)
  CONV_CASE(32, 1,1,0,4)
  CONV_CASE(64, 3,1,1,4)
  CONV_CASE(64, 2,2,1,4)
  CONV_CASE(64, 1,1,0,4)
  CONV_CASE(128,3,1,1,4)
  #undef CONV_CASE
}

static inline void conv_bn(hipStream_t s, size_t inoff, size_t buf, CBN p,
                           int Cin, int Cout, int ks, int stride, int pad,
                           int Din,int Hin,int Win, int Dout,int Hout,int Wout,
                           size_t moff, size_t nsoff, int relu) {
  const int nout = BATCH*Dout*Hout*Wout;
  launch_conv(s, inoff, buf, p.w, Cin, Cout, ks, stride, pad,
              Din,Hin,Win, Dout,Hout,Wout);
  k_stats<<<Cout, 1024, 0, s>>>(buf, moff, nout, Cout);
  size_t total4 = (size_t)nout*(Cout/4);
  k_bn<<<(unsigned)((total4 + 255)/256), 256, 0, s>>>(buf, buf, moff, nsoff, p.g, p.b,
                                                      nout, Cout, relu);
}

extern "C" void kernel_launch(void* const* d_in, const int* in_sizes, int n_in,
                              void* d_out, int out_size, void* d_ws, size_t ws_size,
                              hipStream_t stream) {
  (void)d_ws; (void)ws_size; (void)n_in; (void)out_size;
  const float* feats = (const float*)d_in[0];
  const int*   coors = (const int*)d_in[1];
  const int    nvox  = in_sizes[0] / 4;
  float* out = (float*)d_out;

  // Param leaves: jax pytree order (dict keys sorted alphabetically at every level).
  auto P = [&](int i) { return (const float*)d_in[3 + i]; };
  CBN xc1[3], xc2[3];
  for (int e = 0; e < 3; ++e) {
    xc1[e] = { P(e*6+2), P(e*6+1), P(e*6+0) };
    xc2[e] = { P(e*6+5), P(e*6+4), P(e*6+3) };
  }
  const float* final_b = P(18);
  const float* final_w = P(19);
  CBN in1 = { P(22), P(21), P(20) };
  CBN in2 = { P(25), P(24), P(23) };
  struct Stage { CBN b0c1, b0c2, b0ds, b1c1, b1c2, down; };
  auto mkStage = [&](int s) {
    Stage st;
    st.b0c1 = { P(s+2),  P(s+1),  P(s+0)  };
    st.b0c2 = { P(s+5),  P(s+4),  P(s+3)  };
    st.b0ds = { P(s+8),  P(s+7),  P(s+6)  };
    st.b1c1 = { P(s+11), P(s+10), P(s+9)  };
    st.b1c2 = { P(s+14), P(s+13), P(s+12) };
    st.down = { P(s+17), P(s+16), P(s+15) };
    return st;
  };
  Stage s1 = mkStage(26), s2 = mkStage(44), s3 = mkStage(62);

  const size_t NS0 = OFF_NS+0, NS1 = OFF_NS+1, NS2 = OFF_NS+2, NS3 = OFF_NS+3;

  // ---- scatter to dense ----
  k_zero4<<<4096, 256, 0, stream>>>(OFF_X0, (N0*5)/4);  // X0 (4ch) + M0 contiguous
  k_scatter<<<(nvox + 255)/256, 256, 0, stream>>>(feats, coors, nvox);
  k_masksum<<<1, 1024, 0, stream>>>(OFF_M0, (int)N0, NS0);

  // ---- stem ----
  conv_bn(stream, OFF_X0,  OFF_T0A, in1, 4, 16, 3,1,1, D0,H0,W0, D0,H0,W0, OFF_M0, NS0, 1);
  conv_bn(stream, OFF_T0A, OFF_T0B, in2, 16,16, 3,1,1, D0,H0,W0, D0,H0,W0, OFF_M0, NS0, 1);

  // ---- stage 1 (L0 -> L1, 16 -> 32) ----
  k_downmask<<<(unsigned)((N1+255)/256), 256, 0, stream>>>(OFF_M0, D0,H0,W0, OFF_M1, D1,H1,W1);
  k_masksum<<<1, 1024, 0, stream>>>(OFF_M1, (int)N1, NS1);
  conv_bn(stream, OFF_T0B, OFF_T1A, s1.down, 16,16, 2,2,1, D0,H0,W0, D1,H1,W1, OFF_M1, NS1, 1);
  conv_bn(stream, OFF_T1A, OFF_T1B, s1.b0c1, 16,32, 3,1,1, D1,H1,W1, D1,H1,W1, OFF_M1, NS1, 1);
  conv_bn(stream, OFF_T1B, OFF_T1C, s1.b0c2, 32,32, 3,1,1, D1,H1,W1, D1,H1,W1, OFF_M1, NS1, 0);
  conv_bn(stream, OFF_T1A, OFF_T1D, s1.b0ds, 16,32, 1,1,0, D1,H1,W1, D1,H1,W1, OFF_M1, NS1, 0);
  k_addrelu4<<<2048, 256, 0, stream>>>(OFF_T1C, OFF_T1D, OFF_T1A, (N1*32)/4);
  conv_bn(stream, OFF_T1A, OFF_T1B, s1.b1c1, 32,32, 3,1,1, D1,H1,W1, D1,H1,W1, OFF_M1, NS1, 1);
  conv_bn(stream, OFF_T1B, OFF_T1C, s1.b1c2, 32,32, 3,1,1, D1,H1,W1, D1,H1,W1, OFF_M1, NS1, 0);
  k_addrelu4<<<2048, 256, 0, stream>>>(OFF_T1C, OFF_T1A, OFF_T1B, (N1*32)/4);

  // ---- stage 2 (L1 -> L2, 32 -> 64) ----
  k_downmask<<<(unsigned)((N2+255)/256), 256, 0, stream>>>(OFF_M1, D1,H1,W1, OFF_M2, D2,H2,W2);
  k_masksum<<<1, 1024, 0, stream>>>(OFF_M2, (int)N2, NS2);
  conv_bn(stream, OFF_T1B, OFF_T2A, s2.down, 32,32, 2,2,1, D1,H1,W1, D2,H2,W2, OFF_M2, NS2, 1);
  conv_bn(stream, OFF_T2A, OFF_T2B, s2.b0c1, 32,64, 3,1,1, D2,H2,W2, D2,H2,W2, OFF_M2, NS2, 1);
  conv_bn(stream, OFF_T2B, OFF_T2C, s2.b0c2, 64,64, 3,1,1, D2,H2,W2, D2,H2,W2, OFF_M2, NS2, 0);
  conv_bn(stream, OFF_T2A, OFF_T2D, s2.b0ds, 32,64, 1,1,0, D2,H2,W2, D2,H2,W2, OFF_M2, NS2, 0);
  k_addrelu4<<<1024, 256, 0, stream>>>(OFF_T2C, OFF_T2D, OFF_T2A, (N2*64)/4);
  conv_bn(stream, OFF_T2A, OFF_T2B, s2.b1c1, 64,64, 3,1,1, D2,H2,W2, D2,H2,W2, OFF_M2, NS2, 1);
  conv_bn(stream, OFF_T2B, OFF_T2C, s2.b1c2, 64,64, 3,1,1, D2,H2,W2, D2,H2,W2, OFF_M2, NS2, 0);
  k_addrelu4<<<1024, 256, 0, stream>>>(OFF_T2C, OFF_T2A, OFF_T2B, (N2*64)/4);

  // ---- stage 3 (L2 -> L3, 64 -> 128) ----
  k_downmask<<<(unsigned)((N3+255)/256), 256, 0, stream>>>(OFF_M2, D2,H2,W2, OFF_M3, D3,H3,W3);
  k_masksum<<<1, 1024, 0, stream>>>(OFF_M3, (int)N3, NS3);
  conv_bn(stream, OFF_T2B, OFF_T3A, s3.down, 64,64, 2,2,1, D2,H2,W2, D3,H3,W3, OFF_M3, NS3, 1);
  conv_bn(stream, OFF_T3A, OFF_T3B, s3.b0c1, 64,128, 3,1,1, D3,H3,W3, D3,H3,W3, OFF_M3, NS3, 1);
  conv_bn(stream, OFF_T3B, OFF_T3C, s3.b0c2, 128,128,3,1,1, D3,H3,W3, D3,H3,W3, OFF_M3, NS3, 0);
  conv_bn(stream, OFF_T3A, OFF_T3D, s3.b0ds, 64,128, 1,1,0, D3,H3,W3, D3,H3,W3, OFF_M3, NS3, 0);
  k_addrelu4<<<512, 256, 0, stream>>>(OFF_T3C, OFF_T3D, OFF_T3A, (N3*128)/4);
  conv_bn(stream, OFF_T3A, OFF_T3B, s3.b1c1, 128,128,3,1,1, D3,H3,W3, D3,H3,W3, OFF_M3, NS3, 1);
  conv_bn(stream, OFF_T3B, OFF_T3C, s3.b1c2, 128,128,3,1,1, D3,H3,W3, D3,H3,W3, OFF_M3, NS3, 0);
  k_addrelu4<<<512, 256, 0, stream>>>(OFF_T3C, OFF_T3A, OFF_T3B, (N3*128)/4);

  // ---- extra res blocks (128 -> 128), ping-pong T3B <-> T3A ----
  size_t x = OFF_T3B, y = OFF_T3A;
  for (int e = 0; e < 3; ++e) {
    conv_bn(stream, x, OFF_T3C, xc1[e], 128,128,3,1,1, D3,H3,W3, D3,H3,W3, OFF_M3, NS3, 1);
    conv_bn(stream, OFF_T3C, OFF_T3D, xc2[e], 128,128,3,1,1, D3,H3,W3, D3,H3,W3, OFF_M3, NS3, 0);
    k_addrelu4<<<512, 256, 0, stream>>>(OFF_T3D, x, y, (N3*128)/4);
    size_t tmp = x; x = y; y = tmp;
  }
  // result in `x` (= T3A); T3C/T3D now free

  // ---- final einsum + bias -> d_out (2,256,25,25) ----
  {
    const int NM = BATCH*H3*W3;                 // 1250
    k_repack_final<<<(NM*640 + 255)/256, 256, 0, stream>>>(x, OFF_T3D);
    k_pack_final<<<dim3(COUT_FINAL/16, 20), 32, 0, stream>>>(final_w, 640, 20);
    dim3 grid((NM + 127)/128, COUT_FINAL/(16*4));   // (10, 4), NT=4
    k_final_wmma<<<grid, 256, 0, stream>>>(OFF_T3D, final_b, out);
  }
}
